// CGCNN_89343909691595
// MI455X (gfx1250) — compile-verified, matrix operand order checked by reading, not run on hardware
//
#include <hip/hip_runtime.h>
#include <hip/hip_bf16.h>

#define NN 500000
#define NE 16000000
#define IND 5
#define HID 64
#define NG 512
#define PSTRIDE 24   // padded per-node projection row (floats): 16B-aligned rows

typedef __attribute__((ext_vector_type(16))) _Float16 v16h;
typedef __attribute__((ext_vector_type(8)))  float    v8f;

// ---------------- node init: copy x into working buffer, zero deg ----------------
__global__ void cg_init(const float* __restrict__ x, float* __restrict__ xbuf,
                        float* __restrict__ deg) {
    int n = blockIdx.x * blockDim.x + threadIdx.x;
    if (n >= NN) return;
#pragma unroll
    for (int r = 0; r < IND; ++r) xbuf[n * IND + r] = x[n * IND + r];
    deg[n] = 0.0f;
}

// ---------------- degree: segment_sum(ones, dst) ----------------
__global__ void cg_deg(const int* __restrict__ ei, float* __restrict__ deg) {
    int i = blockIdx.x * blockDim.x + threadIdx.x;
    if (i >= NE) return;
    atomicAdd(&deg[ei[NE + i]], 1.0f);
}

// ---------------- per-node projections: fold z@W.T into node-side terms ----------------
// P[n] layout (stride 24 floats): [0..4]=x·Wf[:,0:5]+bf  [5..9]=x·Ws[:,0:5]+bs
//                                 [12..16]=x·Wf[:,5:10]  [17..21]=x·Ws[:,5:10]
__global__ void cg_proj(const float* __restrict__ xbuf,
                        const float* __restrict__ Wf, const float* __restrict__ bf,
                        const float* __restrict__ Ws, const float* __restrict__ bs,
                        float* __restrict__ P, float* __restrict__ agg) {
    int n = blockIdx.x * blockDim.x + threadIdx.x;
    if (n >= NN) return;
    float xv[IND];
#pragma unroll
    for (int k = 0; k < IND; ++k) xv[k] = xbuf[n * IND + k];
    float* row = P + (size_t)n * PSTRIDE;
#pragma unroll
    for (int r = 0; r < IND; ++r) {
        float pdf = bf[r], pds = bs[r], psf = 0.0f, pss = 0.0f;
#pragma unroll
        for (int k = 0; k < IND; ++k) {
            pdf = fmaf(xv[k], Wf[r * 11 + k], pdf);
            pds = fmaf(xv[k], Ws[r * 11 + k], pds);
            psf = fmaf(xv[k], Wf[r * 11 + 5 + k], psf);
            pss = fmaf(xv[k], Ws[r * 11 + 5 + k], pss);
        }
        row[r]      = pdf;
        row[5 + r]  = pds;
        row[12 + r] = psf;
        row[17 + r] = pss;
        agg[n * IND + r] = 0.0f;  // zero accumulator for this layer
    }
}

// ---------------- edge kernel: gates + scatter (HBM-stream + atomic bound) ----------------
__global__ void cg_edge(const int* __restrict__ ei, const float* __restrict__ ea,
                        const float* __restrict__ Wf, const float* __restrict__ Ws,
                        const float* __restrict__ P, float* __restrict__ agg) {
    int i = blockIdx.x * blockDim.x + threadIdx.x;
    if (i >= NE) return;
    // stream-ahead prefetch of the HBM-resident edge arrays (global_prefetch_b8)
    __builtin_prefetch(ei + i + 16384, 0, 1);
    __builtin_prefetch(ei + NE + i + 16384, 0, 1);
    __builtin_prefetch(ea + i + 16384, 0, 1);

    int src = ei[i];
    int dst = ei[NE + i];
    float e = ea[i];

    const float* rd = P + (size_t)dst * PSTRIDE;        // dst-side terms (+bias)
    const float* rs = P + (size_t)src * PSTRIDE + 12;   // src-side terms
    float pd[10], ps[10];
    *(float4*)&pd[0] = *(const float4*)(rd);
    *(float4*)&pd[4] = *(const float4*)(rd + 4);
    *(float2*)&pd[8] = *(const float2*)(rd + 8);
    *(float4*)&ps[0] = *(const float4*)(rs);
    *(float4*)&ps[4] = *(const float4*)(rs + 4);
    *(float2*)&ps[8] = *(const float2*)(rs + 8);

    float* ao = agg + (size_t)dst * IND;
#pragma unroll
    for (int r = 0; r < IND; ++r) {
        float pf = pd[r]     + ps[r]     + e * Wf[r * 11 + 10];
        float pv = pd[5 + r] + ps[5 + r] + e * Ws[r * 11 + 10];
        float sig = 1.0f / (1.0f + __expf(-pf));                         // sigmoid
        float sp  = fmaxf(pv, 0.0f) + __logf(1.0f + __expf(-fabsf(pv))); // stable softplus
        atomicAdd(&ao[r], sig * sp);
    }
}

// ---------------- node update: x += agg / max(deg,1) ----------------
__global__ void cg_update(float* __restrict__ xbuf, const float* __restrict__ agg,
                          const float* __restrict__ deg) {
    int n = blockIdx.x * blockDim.x + threadIdx.x;
    if (n >= NN) return;
    float inv = 1.0f / fmaxf(deg[n], 1.0f);
#pragma unroll
    for (int r = 0; r < IND; ++r)
        xbuf[n * IND + r] += agg[n * IND + r] * inv;
}

// ---------------- pooling ----------------
__global__ void cg_poolzero(float* __restrict__ gsum, float* __restrict__ gcnt) {
    int i = blockIdx.x * blockDim.x + threadIdx.x;
    if (i < NG * IND) gsum[i] = 0.0f;
    if (i < NG)       gcnt[i] = 0.0f;
}

__global__ void cg_pool(const float* __restrict__ xbuf, const int* __restrict__ batch,
                        float* __restrict__ gsum, float* __restrict__ gcnt) {
    int n = blockIdx.x * blockDim.x + threadIdx.x;
    if (n >= NN) return;
    int g = batch[n];
#pragma unroll
    for (int r = 0; r < IND; ++r)
        atomicAdd(&gsum[g * IND + r], xbuf[n * IND + r]);
    atomicAdd(&gcnt[g], 1.0f);
}

// ---------------- MLP head with WMMA: pooled[512,5] @ W1.T -> relu -> matvec W2 ----------------
// One wave per 16-row tile. A: rows in lanes 0-15 (K=0..4 used, rest 0).
// 4 N-tiles of 16 hidden units each -> 4 v_wmma_f32_16x16x32_f16.
// C layout: VGPR r: lanes 0-15 = (row r, col lane), lanes 16-31 = (row r+8, col lane-16).
__global__ void __launch_bounds__(32)
cg_mlp(const float* __restrict__ gsum, const float* __restrict__ gcnt,
       const float* __restrict__ W1, const float* __restrict__ b1,
       const float* __restrict__ W2, const float* __restrict__ b2,
       float* __restrict__ out) {
    int lane = threadIdx.x;
    int half = lane >> 4;
    int nl   = lane & 15;
    int m0   = blockIdx.x * 16;

    v16h a = {};
    if (lane < 16) {  // reconverges before WMMA; EXEC all-1 at the matrix op
        int m = m0 + lane;
        float inv = 1.0f / fmaxf(gcnt[m], 1.0f);
#pragma unroll
        for (int k = 0; k < IND; ++k)
            a[k] = (_Float16)(gsum[m * IND + k] * inv);
    }

    float acc[8];
#pragma unroll
    for (int r = 0; r < 8; ++r) acc[r] = 0.0f;

#pragma unroll
    for (int nt = 0; nt < 4; ++nt) {
        v16h bm = {};
        if (!half) {  // lanes 16-31 carry K=8..15,24..31 which is all zero-pad
            int n = nt * 16 + nl;  // hidden unit = column
#pragma unroll
            for (int k = 0; k < IND; ++k)
                bm[k] = (_Float16)W1[n * IND + k];
        }
        v8f c = {};
        c = __builtin_amdgcn_wmma_f32_16x16x32_f16(
                false, a, false, bm, (short)0, c, false, false);

        int n = nt * 16 + nl;
        float w2 = W2[n];
        float bb = b1[n];
#pragma unroll
        for (int r = 0; r < 8; ++r) {
            float h = fmaxf(c[r] + bb, 0.0f);   // relu(pooled@W1.T + b1)
            acc[r] += h * w2;                    // fold matvec with W2 (N=1: not a WMMA job)
        }
    }
    // reduce each row across its 16-lane half (xor masks 1..8 keep halves separate)
#pragma unroll
    for (int r = 0; r < 8; ++r) {
        float v = acc[r];
        v += __shfl_xor(v, 1, 32);
        v += __shfl_xor(v, 2, 32);
        v += __shfl_xor(v, 4, 32);
        v += __shfl_xor(v, 8, 32);
        if (nl == 0) out[m0 + r + half * 8] = v + b2[0];
    }
}

// ---------------- host-side orchestration ----------------
static inline size_t alignup(size_t x) { return (x + 255) & ~(size_t)255; }

extern "C" void kernel_launch(void* const* d_in, const int* in_sizes, int n_in,
                              void* d_out, int out_size, void* d_ws, size_t ws_size,
                              hipStream_t stream) {
    const float* x     = (const float*)d_in[0];
    const int*   ei    = (const int*)d_in[1];   // (2, NE): row0 = src, row1 = dst
    const float* ea    = (const float*)d_in[2];
    const int*   batch = (const int*)d_in[3];
    const float* Wf[3] = {(const float*)d_in[4],  (const float*)d_in[8],  (const float*)d_in[12]};
    const float* bf[3] = {(const float*)d_in[5],  (const float*)d_in[9],  (const float*)d_in[13]};
    const float* Ws[3] = {(const float*)d_in[6],  (const float*)d_in[10], (const float*)d_in[14]};
    const float* bs[3] = {(const float*)d_in[7],  (const float*)d_in[11], (const float*)d_in[15]};
    const float* W1 = (const float*)d_in[16];
    const float* b1 = (const float*)d_in[17];
    const float* W2 = (const float*)d_in[18];
    const float* b2 = (const float*)d_in[19];
    float* out = (float*)d_out;

    char* ws = (char*)d_ws;
    size_t o = 0;
    float* xbuf = (float*)(ws + o); o += alignup(sizeof(float) * (size_t)NN * IND);
    float* P    = (float*)(ws + o); o += alignup(sizeof(float) * (size_t)NN * PSTRIDE);
    float* agg  = (float*)(ws + o); o += alignup(sizeof(float) * (size_t)NN * IND);
    float* deg  = (float*)(ws + o); o += alignup(sizeof(float) * (size_t)NN);
    float* gsum = (float*)(ws + o); o += alignup(sizeof(float) * (size_t)NG * IND);
    float* gcnt = (float*)(ws + o); o += alignup(sizeof(float) * (size_t)NG);
    (void)ws_size; (void)in_sizes; (void)n_in; (void)out_size;

    const int NB = 256;
    const int nodeBlocks = (NN + NB - 1) / NB;
    const int edgeBlocks = (NE + NB - 1) / NB;

    cg_init<<<nodeBlocks, NB, 0, stream>>>(x, xbuf, deg);
    cg_deg<<<edgeBlocks, NB, 0, stream>>>(ei, deg);

    for (int l = 0; l < 3; ++l) {
        cg_proj<<<nodeBlocks, NB, 0, stream>>>(xbuf, Wf[l], bf[l], Ws[l], bs[l], P, agg);
        cg_edge<<<edgeBlocks, NB, 0, stream>>>(ei, ea, Wf[l], Ws[l], P, agg);
        cg_update<<<nodeBlocks, NB, 0, stream>>>(xbuf, agg, deg);
    }

    cg_poolzero<<<(NG * IND + NB - 1) / NB, NB, 0, stream>>>(gsum, gcnt);
    cg_pool<<<nodeBlocks, NB, 0, stream>>>(xbuf, batch, gsum, gcnt);

    cg_mlp<<<NG / 16, 32, 0, stream>>>(gsum, gcnt, W1, b1, W2, b2, out);
}
